// PFNLayerV11_44092134261315
// MI455X (gfx1250) — compile-verified, matrix operand order checked by reading, not run on hardware
//
#include <hip/hip_runtime.h>
#include <hip/hip_bf16.h>

typedef __attribute__((ext_vector_type(16))) __bf16 v16bf;
typedef __attribute__((ext_vector_type(8)))  float  v8f;

#define AWF 68  // A tile LDS row stride in floats (4m mod 64 banks -> conflict-free b64 reads)

union ABFrag  { v16bf v; unsigned u[8]; };
union BFrag4  { v16bf v; uint4 q[2]; };

__device__ __forceinline__ unsigned short f2bf(float f) {
    __bf16 h = (__bf16)f;
    return __builtin_bit_cast(unsigned short, h);
}
__device__ __forceinline__ unsigned pack2bf(float lo, float hi) {
    return (unsigned)f2bf(lo) | ((unsigned)f2bf(hi) << 16);
}

// order-preserving float<->uint mapping for atomic max on signed floats
__device__ __forceinline__ unsigned fmap(float f) {
    unsigned b = __float_as_uint(f);
    return (b & 0x80000000u) ? ~b : (b | 0x80000000u);
}
__device__ __forceinline__ float funmap(unsigned u) {
    unsigned b = (u & 0x80000000u) ? (u ^ 0x80000000u) : ~u;
    return __uint_as_float(b);
}
#define WMAX_INIT 0x007FFFFFu   // fmap(-inf)

// generic LDS pointer -> 32-bit LDS byte offset (flat aperture: low 32 bits)
__device__ __forceinline__ unsigned lds_off(const void* p) {
    return (unsigned)(unsigned long long)p;
}

// CDNA5 K index for fragment dword i (16-bit A 16x32 layout; B symmetric)
__device__ __forceinline__ int frag_kb(int i, int kstep, int lane) {
    return kstep * 32 + ((i & 4) ? 16 : 0) + ((lane & 16) ? 8 : 0) + 2 * (i & 3);
}

__device__ __forceinline__ void stat_reduce(float s, float q, int col, int lane,
                                            float* __restrict__ S, float* __restrict__ Q) {
    s += __shfl_xor(s, 16, 32);   // combine the two lane-halves (wave32)
    q += __shfl_xor(q, 16, 32);
    if (lane < 16) { atomicAdd(&S[col], s); atomicAdd(&Q[col], q); }
}

// -------- shared WMMA GEMM core: [64 rows/block, K=64] x [64, NB] --------
// A staged fp32 via async global->LDS DMA; BN(+ReLU) folded into the one-time
// bf16 fragment build. B + biases + inv staged/packed in LDS.
// epi(col, lane, mbase, full, invl, c, bias)
template <int NB, bool XFORM, bool RELU, class EpiF>
__device__ __forceinline__ void gemm_dev(const float* __restrict__ Asrc, long astride, int Mrows,
                                         const float* __restrict__ B0, const float* __restrict__ B1,
                                         const float* __restrict__ B2,
                                         const float* __restrict__ bias0, const float* __restrict__ bias1,
                                         const float* __restrict__ bias2,
                                         const int* __restrict__ inv,
                                         const float* __restrict__ mean, const float* __restrict__ istd,
                                         const float* __restrict__ gg, const float* __restrict__ bb,
                                         EpiF epi) {
    __shared__ float    Afp[64 * AWF];     // raw fp32 A tile
    __shared__ unsigned Bpk[NB * 32];      // B packed in WMMA fragment order
    __shared__ float    BiasL[NB];
    __shared__ int      invL[64];
    __shared__ float    Ascale[64], Ashift[64];
    const int tid = threadIdx.x, lane = tid & 31, wave = tid >> 5;
    const int row0 = blockIdx.x * 64;

    // --- async DMA: A tile (64 rows x 16B chunks) global -> LDS ---
    for (int t = tid; t < 64 * 16; t += 128) {
        int m = t >> 4, c4 = t & 15;
        int r = row0 + m;
        float* dst = &Afp[m * AWF + c4 * 4];
        if (r < Mrows) {
            unsigned doff = lds_off(dst);
            const float* src = Asrc + (long)r * astride + c4 * 4;
            asm volatile("global_load_async_to_lds_b128 %0, %1, off"
                         :: "v"(doff), "v"(src) : "memory");
        } else {
            dst[0] = 0.f; dst[1] = 0.f; dst[2] = 0.f; dst[3] = 0.f;
        }
    }

    // --- stage per-column BN scale/shift, biases, inv ---
    if (XFORM) {
        for (int i = tid; i < 64; i += 128) {
            float s = istd[i] * gg[i];
            Ascale[i] = s;
            Ashift[i] = bb[i] - mean[i] * s;
        }
    }
    const float* Bias_s[3] = {bias0, (NB > 64 ? bias1 : bias0), (NB > 64 ? bias2 : bias0)};
    for (int i = tid; i < NB; i += 128) BiasL[i] = Bias_s[i >> 6][i & 63];
    if (inv) {
        for (int i = tid; i < 64; i += 128) {
            int r = row0 + i;
            invL[i] = (r < Mrows) ? inv[r] : 0;
        }
    }

    // --- pack B into WMMA fragment order: dword idx = ((nt*2+ks)*32+lane)*8+i ---
    const float* Bs[3] = {B0, (NB > 64 ? B1 : B0), (NB > 64 ? B2 : B0)};
    for (int idx = tid; idx < NB * 32; idx += 128) {
        int i  = idx & 7;
        int ln = (idx >> 3) & 31;
        int ks = (idx >> 8) & 1;
        int nt = idx >> 9;
        int kb = frag_kb(i, ks, ln);
        int col = nt * 16 + (ln & 15);
        const float* W = Bs[col >> 6];
        int c = col & 63;
        Bpk[idx] = pack2bf(W[kb * 64 + c], W[(kb + 1) * 64 + c]);
    }

    asm volatile("s_wait_asynccnt 0x0" ::: "memory");
    __syncthreads();

    // --- build A fragments once per wave (fp32 LDS -> transform -> bf16 pack) ---
    const int m = wave * 16 + (lane & 15);
    ABFrag a0, a1;
#pragma unroll
    for (int ks = 0; ks < 2; ++ks) {
        ABFrag& a = ks ? a1 : a0;
#pragma unroll
        for (int i = 0; i < 8; ++i) {
            int kb = frag_kb(i, ks, lane);
            const float2* p = (const float2*)&Afp[m * AWF + kb];
            float x0 = p->x, x1 = p->y;
            if (XFORM) {
                x0 = fmaf(x0, Ascale[kb], Ashift[kb]);
                x1 = fmaf(x1, Ascale[kb + 1], Ashift[kb + 1]);
                if (RELU) { x0 = fmaxf(x0, 0.f); x1 = fmaxf(x1, 0.f); }
            }
            a.u[i] = pack2bf(x0, x1);
        }
    }

    const int  mbase = row0 + wave * 16;
    const bool full  = (mbase + 16 <= Mrows);   // wave-uniform: tail guard only in last block
    const int* invl  = invL + wave * 16;

#pragma unroll 1
    for (int nt = 0; nt < NB / 16; ++nt) {
        BFrag4 b0, b1;
        const uint4* p0 = (const uint4*)&Bpk[((nt * 2 + 0) * 32 + lane) * 8];
        const uint4* p1 = (const uint4*)&Bpk[((nt * 2 + 1) * 32 + lane) * 8];
        b0.q[0] = p0[0]; b0.q[1] = p0[1];
        b1.q[0] = p1[0]; b1.q[1] = p1[1];
        v8f c = {};
        c = __builtin_amdgcn_wmma_f32_16x16x32_bf16(false, a0.v, false, b0.v, (short)0, c, false, false);
        c = __builtin_amdgcn_wmma_f32_16x16x32_bf16(false, a1.v, false, b1.v, (short)0, c, false, false);
        int col = nt * 16 + (lane & 15);
        epi(col, lane, mbase, full, invl, c, BiasL[col]);
    }
}

// -------- GEMM kernel 1: inputs @ [kv_W | w1_W | pre_W] + biases, + BN stats --------
__global__ __launch_bounds__(128) void k_gemm_fused3(
    const float* __restrict__ X, const float* __restrict__ kvW, const float* __restrict__ kvB,
    const float* __restrict__ w1W, const float* __restrict__ w1B,
    const float* __restrict__ preW, const float* __restrict__ preB,
    float* __restrict__ nbuf, float* __restrict__ ssum, float* __restrict__ ssq, int Nr) {
    auto epi = [=](int col, int lane, int mbase, bool full, const int*, v8f c, float bias) {
        int half = (lane & 16) ? 8 : 0;
        float s = 0.f, q = 0.f;
        if (full) {
            float* o = nbuf + (long)(mbase + half) * 192 + col;
#pragma unroll
            for (int r = 0; r < 8; ++r) {
                float val = c[r] + bias;
                o[(long)r * 192] = val;
                s += val; q += val * val;
            }
        } else {
#pragma unroll
            for (int r = 0; r < 8; ++r) {
                int row = mbase + r + half;
                if (row < Nr) {
                    float val = c[r] + bias;
                    nbuf[(long)row * 192 + col] = val;
                    s += val; q += val * val;
                }
            }
        }
        stat_reduce(s, q, col, lane, ssum, ssq);
    };
    gemm_dev<192, false, false>(X, 64, Nr, kvW, w1W, preW, kvB, w1B, preB,
                                nullptr, nullptr, nullptr, nullptr, nullptr, epi);
}

// -------- GEMM: weight = relu(bn(h)) @ w2 + b2 ; epilogue: segment max(weight) --------
__global__ __launch_bounds__(128) void k_gemm_weight(
    const float* __restrict__ nbuf, const float* __restrict__ w2W, const float* __restrict__ w2B,
    const float* __restrict__ mean, const float* __restrict__ istd,
    const float* __restrict__ g, const float* __restrict__ b,
    const int* __restrict__ inv, float* __restrict__ weight, unsigned* __restrict__ wmaxm, int Nr) {
    auto epi = [=](int col, int lane, int mbase, bool full, const int* invl, v8f c, float bias) {
        int half = (lane & 16) ? 8 : 0;
        if (full) {
            float* o = weight + (long)(mbase + half) * 64 + col;
#pragma unroll
            for (int r = 0; r < 8; ++r) {
                float val = c[r] + bias;
                o[(long)r * 64] = val;
                atomicMax(&wmaxm[invl[r + half] * 64 + col], fmap(val));
            }
        } else {
#pragma unroll
            for (int r = 0; r < 8; ++r) {
                int row = mbase + r + half;
                if (row < Nr) {
                    float val = c[r] + bias;
                    weight[(long)row * 64 + col] = val;
                    atomicMax(&wmaxm[invl[r + half] * 64 + col], fmap(val));
                }
            }
        }
    };
    gemm_dev<64, true, true>(nbuf + 64, 192, Nr, w2W, nullptr, nullptr, w2B, nullptr, nullptr,
                             inv, mean, istd, g, b, epi);
}

// -------- GEMM: pw = bn(p) @ posw + b ; epilogue: fs/fc segment sums --------
__global__ __launch_bounds__(128) void k_gemm_pw(
    const float* __restrict__ nbuf, const float* __restrict__ poswW, const float* __restrict__ poswB,
    const float* __restrict__ mean, const float* __restrict__ istd,
    const float* __restrict__ g, const float* __restrict__ b,
    const int* __restrict__ inv, float* __restrict__ add_s, float* __restrict__ add_c, int Nr) {
    auto epi = [=](int col, int lane, int mbase, bool full, const int* invl, v8f c, float bias) {
        int half = (lane & 16) ? 8 : 0;
        float sc = istd[col] * g[col];
        float sh = b[col] - mean[col] * sc;
#pragma unroll
        for (int r = 0; r < 8; ++r) {
            int row = mbase + r + half;
            if (full || row < Nr) {
                float pw = c[r] + bias;
                float p = fmaf(nbuf[(long)row * 192 + 128 + col], sc, sh);
                int sg = invl[r + half];
                atomicAdd(&add_s[sg * 64 + col], p * sinf(pw));
                atomicAdd(&add_c[sg * 64 + col], p * cosf(pw));
            }
        }
    };
    gemm_dev<64, true, false>(nbuf + 128, 192, Nr, poswW, nullptr, nullptr, poswB, nullptr, nullptr,
                              inv, mean, istd, g, b, epi);
}

// -------- GEMM: fraw = q_max @ pre_W + pre_b  (+ V-batch BN stats) --------
__global__ __launch_bounds__(128) void k_gemm_f(
    const float* __restrict__ qmax, const float* __restrict__ preW, const float* __restrict__ preB,
    float* __restrict__ fraw, float* __restrict__ sfsum, float* __restrict__ sfsq, int Vr) {
    auto epi = [=](int col, int lane, int mbase, bool full, const int*, v8f c, float bias) {
        int half = (lane & 16) ? 8 : 0;
        float s = 0.f, q = 0.f;
#pragma unroll
        for (int r = 0; r < 8; ++r) {
            int row = mbase + r + half;
            if (full || row < Vr) {
                float val = c[r] + bias;
                fraw[(long)row * 64 + col] = val;
                s += val; q += val * val;
            }
        }
        stat_reduce(s, q, col, lane, sfsum, sfsq);
    };
    gemm_dev<64, false, false>(qmax, 64, Vr, preW, nullptr, nullptr, preB, nullptr, nullptr,
                               nullptr, nullptr, nullptr, nullptr, nullptr, epi);
}

// -------- GEMM: mw = bn(fraw) @ posw + b ; epilogue: final = (as+cs)*cs + (ac+cc)*cc --------
__global__ __launch_bounds__(128) void k_gemm_mw(
    const float* __restrict__ fraw, const float* __restrict__ poswW, const float* __restrict__ poswB,
    const float* __restrict__ mean, const float* __restrict__ istd,
    const float* __restrict__ g, const float* __restrict__ b,
    const float* __restrict__ add_s, const float* __restrict__ add_c,
    float* __restrict__ finalb, float* __restrict__ slsum, float* __restrict__ slsq, int Vr) {
    auto epi = [=](int col, int lane, int mbase, bool full, const int*, v8f c, float bias) {
        int half = (lane & 16) ? 8 : 0;
        float sc = istd[col] * g[col];
        float sh = b[col] - mean[col] * sc;
        float s = 0.f, q = 0.f;
#pragma unroll
        for (int r = 0; r < 8; ++r) {
            int row = mbase + r + half;
            if (full || row < Vr) {
                long idx = (long)row * 64 + col;
                float mw = c[r] + bias;
                float f = fmaf(fraw[idx], sc, sh);
                float cs = f * sinf(mw), cc = f * cosf(mw);
                float fin = (add_s[idx] + cs) * cs + (add_c[idx] + cc) * cc;
                finalb[idx] = fin;
                s += fin; q += fin * fin;
            }
        }
        stat_reduce(s, q, col, lane, slsum, slsq);
    };
    gemm_dev<64, true, false>(fraw, 64, Vr, poswW, nullptr, nullptr, poswB, nullptr, nullptr,
                              nullptr, mean, istd, g, b, epi);
}

// -------- small kernels --------
__global__ void k_init(float* qmax, unsigned* wmaxm, float* denom, float* S1,
                       float* adds, float* addc, float* cnt,
                       float* s1, float* q1, float* sf, float* qf, float* sl, float* ql,
                       int VC, int V) {
    int i = blockIdx.x * blockDim.x + threadIdx.x;
    if (i < VC) {
        qmax[i] = 0.f; wmaxm[i] = WMAX_INIT;
        denom[i] = 0.f; S1[i] = 0.f; adds[i] = 0.f; addc[i] = 0.f;
    }
    if (i < V) cnt[i] = 0.f;
    if (i < 192) { s1[i] = 0.f; q1[i] = 0.f; }
    if (i < 64) { sf[i] = 0.f; qf[i] = 0.f; sl[i] = 0.f; ql[i] = 0.f; }
}

__global__ void k_stats(const float* __restrict__ S, const float* __restrict__ Q,
                        float* __restrict__ mean, float* __restrict__ istd,
                        int n, float invcnt, float eps, int mixed) {
    int c = blockIdx.x * blockDim.x + threadIdx.x;
    if (c < n) {
        float m = S[c] * invcnt;
        float var = Q[c] * invcnt - m * m;
        float e = mixed ? ((c >= 64 && c < 128) ? 1e-5f : 1e-3f) : eps;
        mean[c] = m;
        istd[c] = rsqrtf(var + e);
    }
}

// x = relu(bn(v)); segment max via uint atomic (x >= 0)
__global__ void k_xmax(const float* __restrict__ nbuf, const int* __restrict__ inv,
                       const float* __restrict__ mean, const float* __restrict__ istd,
                       const float* __restrict__ g, const float* __restrict__ b,
                       unsigned* __restrict__ qmax_u, long total) {
    long i = (long)blockIdx.x * blockDim.x + threadIdx.x;
    if (i >= total) return;
    int col = (int)(i & 63);
    long row = i >> 6;
    float v = nbuf[row * 192 + col];
    float x = fmaxf((v - mean[col]) * istd[col] * g[col] + b[col], 0.f);
    atomicMax(&qmax_u[(long)inv[row] * 64 + col], __float_as_uint(x));
}

// ex = exp(w - m); denom += ex; S1 += ex*v; cnt += 1
__global__ void k_soft(const float* __restrict__ weight, const float* __restrict__ nbuf,
                       const int* __restrict__ inv, const unsigned* __restrict__ wmaxm,
                       float* __restrict__ denom, float* __restrict__ S1, float* __restrict__ cnt,
                       long total) {
    long i = (long)blockIdx.x * blockDim.x + threadIdx.x;
    if (i >= total) return;
    int col = (int)(i & 63);
    long row = i >> 6;
    int sg = inv[row];
    long sidx = (long)sg * 64 + col;
    float w = weight[row * 64 + col];
    float ex = expf(w - funmap(wmaxm[sidx]));
    atomicAdd(&denom[sidx], ex);
    float v = nbuf[row * 192 + col];
    atomicAdd(&S1[sidx], ex * v);
    if (col == 0) atomicAdd(&cnt[sg], 1.f);
}

// link_feat = relu(bn(final)); out = [(lf + S1/denom/cnt)*0.5 | q_max]
__global__ void k_final(const float* __restrict__ finalb, const float* __restrict__ denom,
                        const float* __restrict__ S1, const float* __restrict__ cnt,
                        const float* __restrict__ qmax,
                        const float* __restrict__ mean, const float* __restrict__ istd,
                        const float* __restrict__ g, const float* __restrict__ b,
                        float* __restrict__ out, long total) {
    long i = (long)blockIdx.x * blockDim.x + threadIdx.x;
    if (i >= total) return;
    int col = (int)(i & 63);
    long v = i >> 6;
    float fin = finalb[i];
    float lf = fmaxf((fin - mean[col]) * istd[col] * g[col] + b[col], 0.f);
    float d = denom[i];
    float wx = (d > 0.f ? S1[i] / d : 0.f) / fmaxf(cnt[v], 1.f);
    out[v * 128 + col] = (lf + wx) * 0.5f;
    out[v * 128 + 64 + col] = qmax[i];
}

extern "C" void kernel_launch(void* const* d_in, const int* in_sizes, int n_in,
                              void* d_out, int out_size, void* d_ws, size_t ws_size,
                              hipStream_t stream) {
    (void)n_in; (void)ws_size;
    const float* inputs   = (const float*)d_in[0];
    const int*   inv      = (const int*)  d_in[1];
    const float* kv_W     = (const float*)d_in[2];
    const float* kv_b     = (const float*)d_in[3];
    const float* norm_g   = (const float*)d_in[4];
    const float* norm_b   = (const float*)d_in[5];
    const float* w1_W     = (const float*)d_in[6];
    const float* w1_bias  = (const float*)d_in[7];
    const float* w_bn_g   = (const float*)d_in[8];
    const float* w_bn_b   = (const float*)d_in[9];
    const float* w2_W     = (const float*)d_in[10];
    const float* w2_bias  = (const float*)d_in[11];
    const float* pre_W    = (const float*)d_in[12];
    const float* pre_bias = (const float*)d_in[13];
    const float* pre_bn_g = (const float*)d_in[14];
    const float* pre_bn_b = (const float*)d_in[15];
    const float* posw_W   = (const float*)d_in[16];
    const float* posw_bias= (const float*)d_in[17];
    const float* link_g   = (const float*)d_in[18];
    const float* link_b   = (const float*)d_in[19];

    const int N = in_sizes[1];
    const int V = out_size / 128;
    const long VC = (long)V * 64;

    // workspace carve
    char* wp = (char*)d_ws;
    auto carve = [&](size_t bytes) { void* p = wp; wp += (bytes + 255) & ~(size_t)255; return p; };
    float*    nbuf   = (float*)   carve((size_t)N * 192 * 4);  // [vraw|hraw|praw]
    float*    weight = (float*)   carve((size_t)N * 64 * 4);
    float*    qmax   = (float*)   carve(VC * 4);               // float bits via uint atomicMax
    unsigned* wmaxm  = (unsigned*)carve(VC * 4);
    float*    denom  = (float*)   carve(VC * 4);
    float*    S1     = (float*)   carve(VC * 4);
    float*    add_s  = (float*)   carve(VC * 4);
    float*    add_c  = (float*)   carve(VC * 4);
    float*    fraw   = (float*)   carve(VC * 4);
    float*    finalb = (float*)   carve(VC * 4);
    float*    cnt    = (float*)   carve((size_t)V * 4);
    float*    s1sum  = (float*)carve(192 * 4); float* s1sq = (float*)carve(192 * 4);
    float*    s1mean = (float*)carve(192 * 4); float* s1istd = (float*)carve(192 * 4);
    float*    sfsum  = (float*)carve(64 * 4);  float* sfsq  = (float*)carve(64 * 4);
    float*    sfmean = (float*)carve(64 * 4);  float* sfistd= (float*)carve(64 * 4);
    float*    slsum  = (float*)carve(64 * 4);  float* slsq  = (float*)carve(64 * 4);
    float*    slmean = (float*)carve(64 * 4);  float* slistd= (float*)carve(64 * 4);

    const int gN = (N + 63) / 64;
    const int gV = (V + 63) / 64;
    const long NC = (long)N * 64;

    k_init<<<(int)((VC + 255) / 256), 256, 0, stream>>>(qmax, wmaxm, denom, S1, add_s, add_c, cnt,
                                                        s1sum, s1sq, sfsum, sfsq, slsum, slsq, (int)VC, V);
    k_gemm_fused3<<<gN, 128, 0, stream>>>(inputs, kv_W, kv_b, w1_W, w1_bias, pre_W, pre_bias,
                                          nbuf, s1sum, s1sq, N);
    k_stats<<<1, 256, 0, stream>>>(s1sum, s1sq, s1mean, s1istd, 192, 1.f / (float)N, 0.f, 1);
    k_gemm_weight<<<gN, 128, 0, stream>>>(nbuf, w2_W, w2_bias, s1mean + 64, s1istd + 64,
                                          w_bn_g, w_bn_b, inv, weight, wmaxm, N);
    k_xmax<<<(int)((NC + 255) / 256), 256, 0, stream>>>(nbuf, inv, s1mean, s1istd, norm_g, norm_b,
                                                        (unsigned*)qmax, NC);
    k_gemm_pw<<<gN, 128, 0, stream>>>(nbuf, posw_W, posw_bias, s1mean + 128, s1istd + 128,
                                      pre_bn_g, pre_bn_b, inv, add_s, add_c, N);
    k_soft<<<(int)((NC + 255) / 256), 256, 0, stream>>>(weight, nbuf, inv, wmaxm, denom, S1, cnt, NC);
    k_gemm_f<<<gV, 128, 0, stream>>>(qmax, pre_W, pre_bias, fraw, sfsum, sfsq, V);
    k_stats<<<1, 256, 0, stream>>>(sfsum, sfsq, sfmean, sfistd, 64, 1.f / (float)V, 1e-3f, 0);
    k_gemm_mw<<<gV, 128, 0, stream>>>(fraw, posw_W, posw_bias, sfmean, sfistd, pre_bn_g, pre_bn_b,
                                      add_s, add_c, finalb, slsum, slsq, V);
    k_stats<<<1, 256, 0, stream>>>(slsum, slsq, slmean, slistd, 64, 1.f / (float)V, 1e-3f, 0);
    k_final<<<(int)((VC + 255) / 256), 256, 0, stream>>>(finalb, denom, S1, cnt, qmax,
                                                         slmean, slistd, link_g, link_b,
                                                         (float*)d_out, VC);
}
